// IMFusion_58591943852189
// MI455X (gfx1250) — matrix operand order repaired and checked by reading, not compiled, not run on hardware
//
#include <hip/hip_runtime.h>
#include <hip/hip_bf16.h>

typedef __attribute__((ext_vector_type(16))) __bf16 v16bf;
typedef __attribute__((ext_vector_type(8)))  float  v8f;

union ABFrag { v16bf v; unsigned u[8]; };

__device__ __forceinline__ unsigned short bf16_of(float f) {
    return __builtin_bit_cast(unsigned short, (__bf16)f);
}
__device__ __forceinline__ unsigned pack_bf16(float a, float b) {
    unsigned ua = bf16_of(a), ub = bf16_of(b);
    return ua | (ub << 16);
}
__device__ __forceinline__ v8f zero8() {
    v8f z;
#pragma unroll
    for (int i = 0; i < 8; ++i) z[i] = 0.f;
    return z;
}
__device__ __forceinline__ v8f wmma_bf16(v16bf a, v16bf b, v8f c) {
    return __builtin_amdgcn_wmma_f32_16x16x32_bf16(false, a, false, b, (short)0, c, false, false);
}

// Fast saturating activations (exp -> v_exp_f32, rcp -> v_rcp_f32).
__device__ __forceinline__ float fast_tanh(float t) {
    float e = __expf(2.f * t);
    return 1.f - 2.f * __builtin_amdgcn_rcpf(e + 1.f);
}
__device__ __forceinline__ float fast_sigmoid(float t) {
    return __builtin_amdgcn_rcpf(1.f + __expf(-t));
}

// Async L2->LDS 16-byte copy (CDNA5 GLOBAL_LOAD_ASYNC_TO_LDS_B128, GVS mode).
// All LDS is dynamic shared, so workgroup LDS offsets == byte offsets into smem.
__device__ __forceinline__ void async_copy16(unsigned lds_off, const void* gbase, unsigned goff) {
    asm volatile("global_load_async_to_lds_b128 %0, %1, %2"
                 :: "v"(lds_off), "v"(goff), "s"(gbase)
                 : "memory");
}
__device__ __forceinline__ void wait_async0() {
    asm volatile("s_wait_asynccnt 0x0" ::: "memory");
}

// Issue one 512x32 bf16 weight k-slab into LDS buffer at byte offset bufoff.
// Slab layout: [n][kk0..31] bf16, row stride 40 bf16 (80 B). Two rows per thread.
__device__ __forceinline__ void issue_slab(const unsigned short* W, int ks, unsigned bufoff,
                                           int tid) {
    const unsigned short* gsrc = W + ks * 32;   // byte base += ks*64
    const unsigned r0 = 2 * tid, r1 = r0 + 1;
#pragma unroll
    for (int j = 0; j < 4; ++j) {
        async_copy16(bufoff + r0 * 80 + j * 16, gsrc, r0 * 1024 + j * 16);
        async_copy16(bufoff + r1 * 80 + j * 16, gsrc, r1 * 1024 + j * 16);
    }
}

// A-matrix fragment (bf16 16x32). Per lane: two contiguous 16B blocks at
// row*stride + kbase + 16*hlf (+0 and +16 bytes). 16B-aligned by construction.
__device__ __forceinline__ v16bf load_a_frag(const unsigned short* lds, int rowbase,
                                             int stride, int kbase, int lane) {
    const int hlf = (lane >> 4) & 1;
    const int row = rowbase + (lane & 15);
    const uint4* p = (const uint4*)(lds + row * stride + kbase + 16 * hlf);
    uint4 lo = p[0], hi = p[1];   // +0B and +16B
    ABFrag f;
    f.u[0] = lo.x; f.u[1] = lo.y; f.u[2] = lo.z; f.u[3] = lo.w;
    f.u[4] = hi.x; f.u[5] = hi.y; f.u[6] = hi.z; f.u[7] = hi.w;
    return f.v;
}

// B-matrix fragment (bf16 32x16) from slab [n][kk], stride 40 bf16. Per lane:
// 32 contiguous bytes at n*stride + 16*hlf -> two ds_load_b128.
__device__ __forceinline__ v16bf load_b_frag_lds(const unsigned short* slab, int colbase,
                                                 int lane) {
    const int hlf = (lane >> 4) & 1;
    const int n = colbase + (lane & 15);
    const uint4* p = (const uint4*)(slab + n * 40 + 16 * hlf);
    uint4 lo = p[0], hi = p[1];
    ABFrag f;
    f.u[0] = lo.x; f.u[1] = lo.y; f.u[2] = lo.z; f.u[3] = lo.w;
    f.u[4] = hi.x; f.u[5] = hi.y; f.u[6] = hi.z; f.u[7] = hi.w;
    return f.v;
}

// Same but straight from a global bf16 weight stored [n][k] with k-stride 512.
__device__ __forceinline__ v16bf load_b_frag_g(const unsigned short* W, int colbase,
                                               int kbase, int lane) {
    const int hlf = (lane >> 4) & 1;
    const int n = colbase + (lane & 15);
    const uint4* p = (const uint4*)(W + (size_t)n * 512 + kbase + 16 * hlf);
    uint4 lo = p[0], hi = p[1];
    ABFrag f;
    f.u[0] = lo.x; f.u[1] = lo.y; f.u[2] = lo.z; f.u[3] = lo.w;
    f.u[4] = hi.x; f.u[5] = hi.y; f.u[6] = hi.z; f.u[7] = hi.w;
    return f.v;
}

// ---------------------------------------------------------------------------
// Prep: Wa1 -> Wa1^T bf16 [n][k]; Wl -> Wy bf16 [g][h], Wi bf16 [g][h]
// ---------------------------------------------------------------------------
__global__ __launch_bounds__(256) void k_prep_weights(
    const float* __restrict__ Wa1, const float* __restrict__ Wl,
    unsigned short* __restrict__ Wa1T, unsigned short* __restrict__ Wy,
    unsigned short* __restrict__ Wi) {
    int i = blockIdx.x * 256 + threadIdx.x;   // 262144 elements
    int n = i >> 9, k = i & 511;
    Wa1T[(size_t)n * 512 + k] = bf16_of(Wa1[(size_t)k * 512 + n]);  // transpose
    Wy[(size_t)n * 512 + k]   = bf16_of(Wl[(size_t)n * 1024 + k]);
    Wi[(size_t)n * 512 + k]   = bf16_of(Wl[(size_t)n * 1024 + 512 + k]);
}

// out[:, 0] = y[:, 0]
__global__ __launch_bounds__(256) void k_copy_s0(const float* __restrict__ y,
                                                 float* __restrict__ out) {
    int i = blockIdx.x * 256 + threadIdx.x;   // 65536 float4
    int b = i >> 13, rem = i & 8191;
    const float4* ys = (const float4*)y;
    float4* os = (float4*)out;
    os[(size_t)b * 1048576 + rem] = ys[(size_t)b * 1048576 + rem];
}

// ---------------------------------------------------------------------------
// Kernel 1: per (b,s) — scores = tanh(x@Wa1+ba1)@wa2, softmax(m), pooled
// LDS: slab double-buffer 2x40960 @0, x tile 33280 @81920, scores/attn @115200
// ---------------------------------------------------------------------------
__global__ __launch_bounds__(256) void k_attn_pool(
    const float* __restrict__ x, const unsigned short* __restrict__ Wa1T,
    const float* __restrict__ ba1, const float* __restrict__ wa2,
    float* __restrict__ pooled_f, unsigned short* __restrict__ pooled_h) {
    extern __shared__ char smem[];
    unsigned short* xl = (unsigned short*)(smem + 81920);   // 32 x 520 bf16
    float* scores = (float*)(smem + 115200);                // 32 f32
    float* attn   = scores + 32;                            // 32 f32

    const int tid = threadIdx.x, lane = tid & 31, wv = tid >> 5;
    const int bs = blockIdx.x;                              // b*128 + s
    const float* xg = x + (size_t)bs * 32 * 512;

    issue_slab(Wa1T, 0, 0, tid);   // prefetch first k-slab, overlaps x staging

    // Stage x tile -> bf16 LDS (8192 float pairs)
#pragma unroll
    for (int i = 0; i < 32; ++i) {
        int p = tid + 256 * i;
        int row = p >> 8, cp = p & 255;
        float2 f = *(const float2*)(xg + row * 512 + 2 * cp);
        *(unsigned*)(xl + row * 520 + 2 * cp) = pack_bf16(f.x, f.y);
    }
    if (tid < 32) scores[tid] = 0.f;

    const int rt = wv & 1;        // row tile (16 rows) of the 32-row output
    const int ctb = wv >> 1;      // wave owns col tiles ctb + 4*i
    v8f acc[8];
#pragma unroll
    for (int i = 0; i < 8; ++i) acc[i] = zero8();

    for (int ks = 0; ks < 16; ++ks) {
        const unsigned cur = (unsigned)(ks & 1) * 40960u;
        wait_async0();         // own slab writes for `cur` have landed
        __syncthreads();       // everyone's reads of the other buffer are done
        if (ks + 1 < 16) issue_slab(Wa1T, ks + 1, 40960u - cur, tid);
        const unsigned short* slab = (const unsigned short*)(smem + cur);

        v16bf a = load_a_frag(xl, rt * 16, 520, ks * 32, lane);
        v16bf bf[8];
#pragma unroll
        for (int i = 0; i < 8; ++i) bf[i] = load_b_frag_lds(slab, (ctb + 4 * i) * 16, lane);
#pragma unroll
        for (int i = 0; i < 8; ++i) acc[i] = wmma_bf16(a, bf[i], acc[i]);
    }

    // Epilogue: +ba1, tanh, *wa2, reduce over columns into per-row scores
    const int hlf = (lane >> 4) & 1;
    float part[8];
#pragma unroll
    for (int v = 0; v < 8; ++v) part[v] = 0.f;
#pragma unroll
    for (int i = 0; i < 8; ++i) {
        int col = (ctb + 4 * i) * 16 + (lane & 15);
        float b1 = ba1[col], w2 = wa2[col];
#pragma unroll
        for (int v = 0; v < 8; ++v) part[v] += fast_tanh(acc[i][v] + b1) * w2;
    }
#pragma unroll
    for (int v = 0; v < 8; ++v)
        atomicAdd(&scores[rt * 16 + 8 * hlf + v], part[v]);
    __syncthreads();

    if (tid == 0) {   // softmax over m = 32
        float mx = -1e30f;
        for (int m = 0; m < 32; ++m) mx = fmaxf(mx, scores[m]);
        float sum = 0.f;
        for (int m = 0; m < 32; ++m) { float e = __expf(scores[m] - mx); attn[m] = e; sum += e; }
        float inv = 1.f / sum;
        for (int m = 0; m < 32; ++m) attn[m] *= inv;
    }
    __syncthreads();

    // pooled[h] = sum_m attn[m] * x[m][h]   (f32 from global for accuracy)
    for (int hh = tid; hh < 512; hh += 256) {
        float a = 0.f;
        for (int m = 0; m < 32; ++m) a += attn[m] * xg[m * 512 + hh];
        pooled_f[(size_t)bs * 512 + hh] = a;
        pooled_h[(size_t)bs * 512 + hh] = bf16_of(a);
    }
}

// ---------------------------------------------------------------------------
// Kernel 2: z = i_x @ Wi^T + bl   (rows r = b*127 + t, t = s-1)
// ---------------------------------------------------------------------------
__global__ __launch_bounds__(256) void k_gate_bias(
    const unsigned short* __restrict__ pooled_h, const unsigned short* __restrict__ Wi,
    const float* __restrict__ bl, float* __restrict__ zt) {
    extern __shared__ char smem[];
    unsigned short* al = (unsigned short*)smem;   // 16 x 520 bf16
    const int tid = threadIdx.x, lane = tid & 31, wv = tid >> 5;
    const int r0 = blockIdx.x * 16;

#pragma unroll
    for (int i = 0; i < 16; ++i) {
        int p = tid + 256 * i;
        int row = p >> 8, cp = p & 255;
        int r = r0 + row;
        unsigned val = 0;
        if (r < 1016) {
            int b = r / 127, t = r - b * 127;
            val = *(const unsigned*)(pooled_h + (size_t)(b * 128 + t) * 512 + 2 * cp);
        }
        *(unsigned*)(al + row * 520 + 2 * cp) = val;
    }
    __syncthreads();

    v8f acc[4];
#pragma unroll
    for (int j = 0; j < 4; ++j) acc[j] = zero8();
    for (int ks = 0; ks < 16; ++ks) {
        v16bf a = load_a_frag(al, 0, 520, ks * 32, lane);
        v16bf bf[4];
#pragma unroll
        for (int j = 0; j < 4; ++j) bf[j] = load_b_frag_g(Wi, (wv * 4 + j) * 16, ks * 32, lane);
#pragma unroll
        for (int j = 0; j < 4; ++j) acc[j] = wmma_bf16(a, bf[j], acc[j]);
    }

    const int hlf = (lane >> 4) & 1;
#pragma unroll
    for (int j = 0; j < 4; ++j) {
        int col = (wv * 4 + j) * 16 + (lane & 15);
        float bb = bl[col];
#pragma unroll
        for (int v = 0; v < 8; ++v) {
            int r = r0 + v + 8 * hlf;
            if (r < 1016) zt[(size_t)r * 512 + col] = acc[j][v] + bb;
        }
    }
}

// ---------------------------------------------------------------------------
// Kernel 3: per (b, s>=1) — pre = y@Wy^T + z; out = y + i_x * sigmoid(pre)
// LDS: slab double-buffer 2x40960 @0, y tile 66560 @81920, z @148480, ix @150528
// ---------------------------------------------------------------------------
__global__ __launch_bounds__(256) void k_main(
    const float* __restrict__ y, const unsigned short* __restrict__ Wy,
    const float* __restrict__ pooled_f, const float* __restrict__ zt,
    float* __restrict__ out) {
    extern __shared__ char smem[];
    unsigned short* yl = (unsigned short*)(smem + 81920);   // 64 x 520 bf16
    float* zr  = (float*)(smem + 148480);                   // 512 f32
    float* ixr = (float*)(smem + 150528);                   // 512 f32

    const int tid = threadIdx.x, lane = tid & 31, wv = tid >> 5;
    const int wg = blockIdx.x;
    const int b = wg / 127, sm1 = wg - b * 127, s = sm1 + 1;
    const float* yg = y + (size_t)(b * 128 + s) * 64 * 512;
    float* og = out + (size_t)(b * 128 + s) * 64 * 512;
    const float* zg = zt + (size_t)(b * 127 + sm1) * 512;
    const float* ixg = pooled_f + (size_t)(b * 128 + sm1) * 512;

    issue_slab(Wy, 0, 0, tid);   // prefetch first k-slab, overlaps y staging

    // Stage y tile (64x512) -> bf16 LDS; z row and i_x row -> f32 LDS
#pragma unroll
    for (int i = 0; i < 64; ++i) {
        int p = tid + 256 * i;
        int row = p >> 8, cp = p & 255;
        float2 f = *(const float2*)(yg + row * 512 + 2 * cp);
        *(unsigned*)(yl + row * 520 + 2 * cp) = pack_bf16(f.x, f.y);
    }
    for (int hh = tid; hh < 512; hh += 256) { zr[hh] = zg[hh]; ixr[hh] = ixg[hh]; }

    v8f acc[16];
#pragma unroll
    for (int i = 0; i < 16; ++i) acc[i] = zero8();

    for (int ks = 0; ks < 16; ++ks) {
        const unsigned cur = (unsigned)(ks & 1) * 40960u;
        wait_async0();
        __syncthreads();
        if (ks + 1 < 16) issue_slab(Wy, ks + 1, 40960u - cur, tid);
        const unsigned short* slab = (const unsigned short*)(smem + cur);

        v16bf a[4];
#pragma unroll
        for (int rt = 0; rt < 4; ++rt) a[rt] = load_a_frag(yl, rt * 16, 520, ks * 32, lane);
        v16bf bf[4];
#pragma unroll
        for (int j = 0; j < 4; ++j) bf[j] = load_b_frag_lds(slab, (wv * 4 + j) * 16, lane);
#pragma unroll
        for (int j = 0; j < 4; ++j)
#pragma unroll
            for (int rt = 0; rt < 4; ++rt)
                acc[rt * 4 + j] = wmma_bf16(a[rt], bf[j], acc[rt * 4 + j]);
    }

    // Epilogue: out = y + i_x * sigmoid(acc + z)
    const int hlf = (lane >> 4) & 1;
#pragma unroll
    for (int rt = 0; rt < 4; ++rt) {
#pragma unroll
        for (int j = 0; j < 4; ++j) {
            int col = (wv * 4 + j) * 16 + (lane & 15);
            float zc = zr[col], ixc = ixr[col];
#pragma unroll
            for (int v = 0; v < 8; ++v) {
                int n = rt * 16 + v + 8 * hlf;
                float pre = acc[rt * 4 + j][v] + zc;
                og[n * 512 + col] = yg[n * 512 + col] + ixc * fast_sigmoid(pre);
            }
        }
    }
}

// ---------------------------------------------------------------------------
extern "C" void kernel_launch(void* const* d_in, const int* in_sizes, int n_in,
                              void* d_out, int out_size, void* d_ws, size_t ws_size,
                              hipStream_t stream) {
    const float* x   = (const float*)d_in[0];   // (8,128,32,512)
    const float* y   = (const float*)d_in[1];   // (8,128,64,512)
    const float* Wa1 = (const float*)d_in[2];   // (512,512)
    const float* ba1 = (const float*)d_in[3];   // (512)
    const float* wa2 = (const float*)d_in[4];   // (512)
    const float* Wl  = (const float*)d_in[5];   // (512,1024)
    const float* bl  = (const float*)d_in[6];   // (512)
    float* out = (float*)d_out;
    char* ws = (char*)d_ws;

    unsigned short* Wa1T     = (unsigned short*)(ws);            // 512 KB
    unsigned short* Wy       = (unsigned short*)(ws + 524288);   // 512 KB
    unsigned short* Wi       = (unsigned short*)(ws + 1048576);  // 512 KB
    float*          pooled_f = (float*)(ws + 1572864);           // 2 MB
    unsigned short* pooled_h = (unsigned short*)(ws + 3670016);  // 1 MB
    float*          zt       = (float*)(ws + 4718592);           // ~2 MB

    k_prep_weights<<<1024, 256, 0, stream>>>(Wa1, Wl, Wa1T, Wy, Wi);
    k_copy_s0<<<256, 256, 0, stream>>>(y, out);
    k_attn_pool<<<1024, 256, 115456, stream>>>(x, Wa1T, ba1, wa2, pooled_f, pooled_h);
    k_gate_bias<<<64, 256, 16640, stream>>>(pooled_h, Wi, bl, zt);
    k_main<<<1016, 256, 152576, stream>>>(y, Wy, pooled_f, zt, out);
}
